// GATGPP_13683765805699
// MI455X (gfx1250) — matrix operand order repaired and checked by reading, not compiled, not run on hardware
//
#include <hip/hip_runtime.h>
#include <cstdint>
#include <cstddef>

typedef __attribute__((ext_vector_type(16))) _Float16 v16h;
typedef __attribute__((ext_vector_type(8)))  float    v8f;

#define DEV_INLINE __device__ __forceinline__

// ---- problem constants (match reference) ----
constexpr int kN   = 50000;       // nodes  (divisible by 16)
constexpr int kFIN = 128;         // input features
constexpr int kH   = 4;           // heads
constexpr int kC   = 64;          // channels/head
constexpr int kHC  = 256;         // H*C
constexpr int kE   = 800000;      // edges
constexpr int kEP  = kE + kN;     // edges + self loops
constexpr int kG   = 128;         // graphs
constexpr int kOUT = 8;           // output dim
constexpr unsigned kEncNegInf = 0x007FFFFFu; // enc(-inf)

// monotonic float<->uint map for atomic fp32 max
DEV_INLINE unsigned enc_f32(float f){ unsigned u=__float_as_uint(f); return (u&0x80000000u)? ~u : (u|0x80000000u); }
DEV_INLINE float    dec_f32(unsigned u){ return (u&0x80000000u)? __uint_as_float(u&0x7FFFFFFFu) : __uint_as_float(~u); }
DEV_INLINE float    lrelu(float v){ return v>0.f ? v : 0.2f*v; }

DEV_INLINE void edge_sd(const int* __restrict__ ei, int e, int& s, int& d){
  if (e < kE){ s = ei[e]; d = ei[kE + e]; } else { s = e - kE; d = e - kE; }
}

// ---------------- utility kernels ----------------
__global__ void k_fill_u32(unsigned* __restrict__ p, unsigned v, int n){
  int i = blockIdx.x*blockDim.x + threadIdx.x;
  if (i < n) p[i] = v;
}

__global__ void k_f32_to_f16(const float* __restrict__ in, _Float16* __restrict__ out, int n){
  int i = blockIdx.x*blockDim.x + threadIdx.x;
  if (i < n) out[i] = (_Float16)in[i];
}

// ---------------- pack weight B[K,Ncol] (f32, row-major) into WMMA fragment order ----
// Bp index: (((tn*(K/32) + ks)*32 + lane)*16 + i)
//   b[i] = B[ks*32 + (i<8 ? 8*half + i : 16 + 8*half + i-8)][tn*16 + (lane&15)], half = lane>>4
__global__ void k_pack_B(const float* __restrict__ B, _Float16* __restrict__ Bp,
                         int K, int Ncol){
  int idx = blockIdx.x*blockDim.x + threadIdx.x;
  int total = K * Ncol;
  if (idx >= total) return;
  int i    =  idx        & 15;
  int lane = (idx >> 4)  & 31;
  int rest =  idx >> 9;
  int ksteps = K >> 5;
  int ks = rest % ksteps;
  int tn = rest / ksteps;
  int half = lane >> 4, l15 = lane & 15;
  int row = (ks << 5) + (half << 3) + (i & 7) + ((i >> 3) << 4);   // K index
  int col = (tn << 4) + l15;               // N index
  Bp[idx] = (_Float16)B[(size_t)row * Ncol + col];
}

// ---------------- WMMA GEMM: C[M,Ncol] = A[M,K](f16) * B(packed f16) ----------------
// one wave per 16x64 output strip; software-pipelined, fully unrolled k-loop.
template<int K, int Ncol>
__global__ void k_gemm_f16_wmma(const _Float16* __restrict__ A,
                                const _Float16* __restrict__ Bp,
                                float* __restrict__ Cmat,
                                int M){
  constexpr int NT     = 4;        // 16-wide col tiles per wave (64 cols)
  constexpr int KSTEPS = K >> 5;
  constexpr int BSTEP  = 32 * 16;  // halves per k-step per tile stream

  const int wavesPerBlock = blockDim.x >> 5;
  int wave = blockIdx.x * wavesPerBlock + (threadIdx.x >> 5);
  constexpr int colGroups = Ncol >> 6;
  int total = (M >> 4) * colGroups;
  if (wave >= total) return;                  // wave-uniform -> EXEC all-1s for WMMA
  int tm = wave / colGroups;
  int cg = wave - tm * colGroups;
  int lane = threadIdx.x & 31;
  int half = lane >> 4;
  int l15  = lane & 15;
  int arow = (tm << 4) + l15;

  v8f acc[NT];
#pragma unroll
  for (int t = 0; t < NT; ++t) acc[t] = (v8f){};

  const _Float16* ap = A + (size_t)arow * K + (half << 3);
  const _Float16* bbase = Bp + ((((size_t)(cg << 2) * KSTEPS) * 32 + lane) << 4);

  // ---- preload k-step 0 ----
  v16h a_cur, b_cur[NT];
#pragma unroll
  for (int i = 0; i < 8; ++i){ a_cur[i] = ap[i]; a_cur[i+8] = ap[16+i]; }
#pragma unroll
  for (int t = 0; t < NT; ++t)
    b_cur[t] = *(const v16h*)(bbase + (size_t)t * KSTEPS * BSTEP);

  // ---- pipelined main loop (fully unrolled, immediate offsets) ----
#pragma unroll
  for (int ks = 0; ks < KSTEPS - 1; ++ks){
    v16h a_nxt, b_nxt[NT];
    const _Float16* apn = ap + (ks + 1) * 32;
#pragma unroll
    for (int i = 0; i < 8; ++i){ a_nxt[i] = apn[i]; a_nxt[i+8] = apn[16+i]; }
#pragma unroll
    for (int t = 0; t < NT; ++t)
      b_nxt[t] = *(const v16h*)(bbase + ((size_t)t * KSTEPS + (ks + 1)) * BSTEP);
#pragma unroll
    for (int t = 0; t < NT; ++t)
      acc[t] = __builtin_amdgcn_wmma_f32_16x16x32_f16(false, a_cur, false, b_cur[t],
                                                      (short)0, acc[t], false, false);
    a_cur = a_nxt;
#pragma unroll
    for (int t = 0; t < NT; ++t) b_cur[t] = b_nxt[t];
  }
  // ---- epilogue k-step ----
#pragma unroll
  for (int t = 0; t < NT; ++t)
    acc[t] = __builtin_amdgcn_wmma_f32_16x16x32_f16(false, a_cur, false, b_cur[t],
                                                    (short)0, acc[t], false, false);

  // C/D layout: VGPR v -> row tm*16 + half*8 + v, col = tile*16 + (lane&15)
  int crow = (tm << 4) + (half << 3);
#pragma unroll
  for (int t = 0; t < NT; ++t){
    int ccol = (((cg << 2) + t) << 4) + l15;
#pragma unroll
    for (int v = 0; v < 8; ++v)
      Cmat[(size_t)(crow + v) * Ncol + ccol] = acc[t][v];
  }
}

// ---------------- alpha_src / alpha_dst : per (node, head) dot over C ----------------
__global__ void k_alpha(const float* __restrict__ hbuf,
                        const float* __restrict__ a_s, const float* __restrict__ a_d,
                        float* __restrict__ asrc, float* __restrict__ adst, int n){
  int idx = blockIdx.x*blockDim.x + threadIdx.x;
  if (idx >= n * kH) return;
  int node = idx >> 2, hh = idx & 3;
  const float* hp = hbuf + (size_t)node * kHC + hh * kC;
  const float* as = a_s + hh * kC;
  const float* ad = a_d + hh * kC;
  float ss = 0.f, sd = 0.f;
#pragma unroll 8
  for (int c = 0; c < kC; ++c){ float v = hp[c]; ss += v * as[c]; sd += v * ad[c]; }
  asrc[idx] = ss; adst[idx] = sd;
}

// ---------------- edge pass 1: segment max of leaky_relu(e) by dst ----------------
__global__ void k_edge_max(const int* __restrict__ ei,
                           const float* __restrict__ asrc, const float* __restrict__ adst,
                           unsigned* __restrict__ m){
  int e = blockIdx.x*blockDim.x + threadIdx.x;
  if (e >= kEP) return;
  int s, d; edge_sd(ei, e, s, d);
#pragma unroll
  for (int hh = 0; hh < kH; ++hh){
    float v = lrelu(asrc[s*kH + hh] + adst[d*kH + hh]);
    atomicMax(&m[d*kH + hh], enc_f32(v));
  }
}

// ---------------- edge pass 2: ex = exp(e - m[dst]); denom[dst] += ex ----------------
__global__ void k_edge_expsum(const int* __restrict__ ei,
                              const float* __restrict__ asrc, const float* __restrict__ adst,
                              const unsigned* __restrict__ m,
                              float* __restrict__ exbuf, float* __restrict__ denom){
  int e = blockIdx.x*blockDim.x + threadIdx.x;
  if (e >= kEP) return;
  int s, d; edge_sd(ei, e, s, d);
#pragma unroll
  for (int hh = 0; hh < kH; ++hh){
    float v  = lrelu(asrc[s*kH + hh] + adst[d*kH + hh]);
    float ex = __expf(v - dec_f32(m[d*kH + hh]));
    exbuf[(size_t)e*kH + hh] = ex;
    atomicAdd(&denom[d*kH + hh], ex);
  }
}

// ---------------- edge pass 3: agg[dst,h,c] += h[src,h,c] * alpha ----------------
__global__ void k_edge_agg(const int* __restrict__ ei,
                           const float* __restrict__ exbuf, const float* __restrict__ denom,
                           const float* __restrict__ hbuf, float* __restrict__ agg){
  long long idx = (long long)blockIdx.x*blockDim.x + threadIdx.x;
  if (idx >= (long long)kEP * kC) return;
  int e = (int)(idx >> 6);
  int c = (int)(idx & 63);
  int s, d; edge_sd(ei, e, s, d);
  const float* hs = hbuf + (size_t)s * kHC;
  float*       ag = agg  + (size_t)d * kHC;
#pragma unroll
  for (int hh = 0; hh < kH; ++hh){
    float w = exbuf[(size_t)e*kH + hh] / (denom[d*kH + hh] + 1e-16f);
    atomicAdd(&ag[hh*kC + c], hs[hh*kC + c] * w);
  }
}

// ---------------- layer1 epilogue: +bias, ELU, convert to f16 for GEMM2 ----------------
__global__ void k_bias_elu_f16(const float* __restrict__ agg, const float* __restrict__ b1,
                               _Float16* __restrict__ acth){
  long long idx = (long long)blockIdx.x*blockDim.x + threadIdx.x;
  if (idx >= (long long)kN * kHC) return;
  int j = (int)(idx & (kHC - 1));
  float v = agg[idx] + b1[j];
  v = v > 0.f ? v : expm1f(v);               // ELU(alpha=1)
  acth[idx] = (_Float16)v;
}

// ---------------- layer2 epilogue: mean over heads + bias -> h2[N,C] ----------------
__global__ void k_mean_bias(const float* __restrict__ agg, const float* __restrict__ b2,
                            float* __restrict__ h2){
  long long idx = (long long)blockIdx.x*blockDim.x + threadIdx.x;
  if (idx >= (long long)kN * kC) return;
  int c = (int)(idx & 63);
  int n = (int)(idx >> 6);
  const float* ag = agg + (size_t)n * kHC;
  float v = 0.25f * (ag[c] + ag[kC + c] + ag[2*kC + c] + ag[3*kC + c]) + b2[c];
  h2[idx] = v;
}

// ---------------- pooling ----------------
__global__ void k_pool_sum(const float* __restrict__ h2, const int* __restrict__ batch,
                           float* __restrict__ pool){
  long long idx = (long long)blockIdx.x*blockDim.x + threadIdx.x;
  if (idx >= (long long)kN * kC) return;
  int c = (int)(idx & 63);
  int n = (int)(idx >> 6);
  atomicAdd(&pool[batch[n]*kC + c], h2[idx]);
}

__global__ void k_pool_cnt(const int* __restrict__ batch, float* __restrict__ cnt){
  int n = blockIdx.x*blockDim.x + threadIdx.x;
  if (n < kN) atomicAdd(&cnt[batch[n]], 1.0f);
}

// ---------------- final head: out[G,OUT] = (pool/cnt) @ fc_w + fc_b ----------------
__global__ void k_final(const float* __restrict__ pool, const float* __restrict__ cnt,
                        const float* __restrict__ fc_w, const float* __restrict__ fc_b,
                        float* __restrict__ out){
  int idx = blockIdx.x*blockDim.x + threadIdx.x;
  if (idx >= kG * kOUT) return;
  int g = idx >> 3, o = idx & 7;
  float inv = 1.0f / fmaxf(cnt[g], 1.0f);
  const float* pg = pool + g * kC;
  float s = 0.f;
#pragma unroll 8
  for (int c = 0; c < kC; ++c) s += (pg[c] * inv) * fc_w[c*kOUT + o];
  out[idx] = s + fc_b[o];
}

// =====================================================================
extern "C" void kernel_launch(void* const* d_in, const int* in_sizes, int n_in,
                              void* d_out, int out_size, void* d_ws, size_t ws_size,
                              hipStream_t stream){
  (void)in_sizes; (void)n_in; (void)out_size;
  const float* x    = (const float*)d_in[0];
  const int*   ei   = (const int*)  d_in[1];   // [2,E]
  const int*   bat  = (const int*)  d_in[2];   // [N]
  const float* W1   = (const float*)d_in[3];
  const float* a1s  = (const float*)d_in[4];
  const float* a1d  = (const float*)d_in[5];
  const float* b1   = (const float*)d_in[6];
  const float* W2   = (const float*)d_in[7];
  const float* a2s  = (const float*)d_in[8];
  const float* a2d  = (const float*)d_in[9];
  const float* b2   = (const float*)d_in[10];
  const float* fcw  = (const float*)d_in[11];
  const float* fcb  = (const float*)d_in[12];
  float* out = (float*)d_out;

  // ---- bump allocator over d_ws ----
  char* base = (char*)d_ws;
  size_t off = 0;
  auto alloc = [&](size_t bytes)->char*{
    char* p = base + off;
    off = (off + bytes + 255) & ~(size_t)255;
    return p;
  };
  float*     hbuf  = (float*)    alloc((size_t)kN*kHC*4);   // GEMM out (reused L1/L2)
  float*     agg   = (float*)    alloc((size_t)kN*kHC*4);   // message agg (reused)
  _Float16*  acth  = (_Float16*) alloc((size_t)kN*kHC*2);   // ELU(h1) f16
  _Float16*  xh    = (_Float16*) alloc((size_t)kN*kFIN*2);  // x f16 ; reused as h2 after GEMM1
  float*     exbuf = (float*)    alloc((size_t)kEP*kH*4);
  _Float16*  W1p   = (_Float16*) alloc((size_t)kFIN*kHC*2); // packed fragments
  _Float16*  W2p   = (_Float16*) alloc((size_t)kHC*kHC*2);  // packed fragments
  float*     asrc  = (float*)    alloc((size_t)kN*kH*4);
  float*     adst  = (float*)    alloc((size_t)kN*kH*4);
  unsigned*  mmax  = (unsigned*) alloc((size_t)kN*kH*4);
  float*     denom = (float*)    alloc((size_t)kN*kH*4);
  float*     pool  = (float*)    alloc((size_t)kG*kC*4);
  float*     cnt   = (float*)    alloc((size_t)kG*4);
  float*     h2    = (float*)xh;                            // reuse (xh dead after GEMM1)
  (void)ws_size;

  const int B = 256;
  auto gr = [](long long n, int b){ return (unsigned)((n + b - 1) / b); };

  // --- convert / pack inputs to f16 ---
  k_f32_to_f16<<<gr((long long)kN*kFIN, B), B, 0, stream>>>(x, xh, kN*kFIN);
  k_pack_B<<<gr(kFIN*kHC, B), B, 0, stream>>>(W1, W1p, kFIN, kHC);
  k_pack_B<<<gr(kHC*kHC, B),  B, 0, stream>>>(W2, W2p, kHC, kHC);

  const int strips = (kN/16) * (kHC/64);        // 3125 * 4 = 12500 waves
  const int gemmBlocks = (strips + 7) / 8;      // 8 waves per 256-thread block

  // ================= Layer 1 =================
  k_gemm_f16_wmma<kFIN, kHC><<<gemmBlocks, B, 0, stream>>>(xh, W1p, hbuf, kN);
  k_alpha<<<gr((long long)kN*kH, B), B, 0, stream>>>(hbuf, a1s, a1d, asrc, adst, kN);
  k_fill_u32<<<gr((long long)kN*kH, B), B, 0, stream>>>(mmax, kEncNegInf, kN*kH);
  k_fill_u32<<<gr((long long)kN*kH, B), B, 0, stream>>>((unsigned*)denom, 0u, kN*kH);
  k_fill_u32<<<gr((long long)kN*kHC, B), B, 0, stream>>>((unsigned*)agg, 0u, kN*kHC);
  k_edge_max   <<<gr(kEP, B), B, 0, stream>>>(ei, asrc, adst, mmax);
  k_edge_expsum<<<gr(kEP, B), B, 0, stream>>>(ei, asrc, adst, mmax, exbuf, denom);
  k_edge_agg   <<<gr((long long)kEP*kC, B), B, 0, stream>>>(ei, exbuf, denom, hbuf, agg);
  k_bias_elu_f16<<<gr((long long)kN*kHC, B), B, 0, stream>>>(agg, b1, acth);

  // ================= Layer 2 =================
  k_gemm_f16_wmma<kHC, kHC><<<gemmBlocks, B, 0, stream>>>(acth, W2p, hbuf, kN);
  k_alpha<<<gr((long long)kN*kH, B), B, 0, stream>>>(hbuf, a2s, a2d, asrc, adst, kN);
  k_fill_u32<<<gr((long long)kN*kH, B), B, 0, stream>>>(mmax, kEncNegInf, kN*kH);
  k_fill_u32<<<gr((long long)kN*kH, B), B, 0, stream>>>((unsigned*)denom, 0u, kN*kH);
  k_fill_u32<<<gr((long long)kN*kHC, B), B, 0, stream>>>((unsigned*)agg, 0u, kN*kHC);
  k_edge_max   <<<gr(kEP, B), B, 0, stream>>>(ei, asrc, adst, mmax);
  k_edge_expsum<<<gr(kEP, B), B, 0, stream>>>(ei, asrc, adst, mmax, exbuf, denom);
  k_edge_agg   <<<gr((long long)kEP*kC, B), B, 0, stream>>>(ei, exbuf, denom, hbuf, agg);
  k_mean_bias  <<<gr((long long)kN*kC, B), B, 0, stream>>>(agg, b2, h2);

  // ================= Pool + head =================
  k_fill_u32<<<gr(kG*kC, B), B, 0, stream>>>((unsigned*)pool, 0u, kG*kC);
  k_fill_u32<<<gr(kG, B),    B, 0, stream>>>((unsigned*)cnt,  0u, kG);
  k_pool_sum<<<gr((long long)kN*kC, B), B, 0, stream>>>(h2, bat, pool);
  k_pool_cnt<<<gr(kN, B), B, 0, stream>>>(bat, cnt);
  k_final<<<gr(kG*kOUT, B), B, 0, stream>>>(pool, cnt, fcw, fcb, out);
}
